// WOQ_Linear_Mul_Add_Model_82841329206006
// MI455X (gfx1250) — matrix operand order repaired and checked by reading, not compiled, not run on hardware
//
#include <hip/hip_runtime.h>
#include <hip/hip_bf16.h>

// WOQ linear + mul/add epilogue for MI455X (gfx1250, wave32, WMMA).
// C[t,o] = sum_k x[t,k] * ((q[o,k]-128)*scale[o]) ; out = C*mul + (bias*mul+add)
// Compute-bound (550 GFLOP vs ~335MB HBM) -> bf16 WMMA 16x16x32.
// Tiling chosen so LDS read traffic (~1KB/WMMA) matches XDL issue rate:
// 256x128 block, 8 waves in 4(M)x2(N) grid, each wave computes 64x64 (4x4 tiles).

typedef __attribute__((ext_vector_type(16))) __bf16 v16bf;
typedef __attribute__((ext_vector_type(8)))  __bf16 v8bf;
typedef __attribute__((ext_vector_type(8)))  float  v8f;

#define TOKENS 16384
#define INF    4096
#define OUTF   4096
#define BM 256
#define BN 128
#define BK 32
#define LDT (BK + 8)   // padded LDS row: 40 bf16 = 80B (16B aligned, conflict-free)

// pack two dequantized bf16 values into one dword
static __device__ __forceinline__ unsigned dq2(int q0, int q1, float s) {
    float f0 = (float)(q0 - 128) * s;
    float f1 = (float)(q1 - 128) * s;
    __bf16 b0 = (__bf16)f0;
    __bf16 b1 = (__bf16)f1;
    unsigned short u0, u1;
    __builtin_memcpy(&u0, &b0, 2);
    __builtin_memcpy(&u1, &b1, 2);
    return (unsigned)u0 | ((unsigned)u1 << 16);
}

static __device__ __forceinline__ v16bf comb(v8bf lo, v8bf hi) {
    v16bf r;
#pragma unroll
    for (int i = 0; i < 8; ++i) { r[i] = lo[i]; r[i + 8] = hi[i]; }
    return r;
}

__global__ __launch_bounds__(256)
void woq_wmma_kernel(const __bf16* __restrict__ x,
                     const int*    __restrict__ qw,
                     const float*  __restrict__ scales,
                     const __bf16* __restrict__ bias,
                     const __bf16* __restrict__ mulv,
                     const __bf16* __restrict__ addv,
                     __bf16* __restrict__ out)
{
    __shared__ __bf16 As[BM * LDT];   // 256 rows x 32 K (bf16)        ~20.5KB
    __shared__ __bf16 Bs[BN * LDT];   // 128 cols x 32 K (dequant bf16) ~10.3KB

    const int tid     = threadIdx.x;
    const int lane    = tid & 31;
    const int wave    = tid >> 5;       // 0..7
    const int laneLow = lane & 15;
    const int laneHi  = lane >> 4;      // 0/1
    const int wm      = wave & 3;       // M strip: 64 rows each (4 strips)
    const int wn      = wave >> 2;      // N strip: 64 cols each (2 strips)

    const int mBase = blockIdx.y * BM;
    const int nBase = blockIdx.x * BN;

    // ---- staging assignments (K-invariant) ----
    // A: each thread copies one full 32-K row (64B, 4 x b128)
    const int   rowA  = tid;            // 0..255
    // B: thread pair per output channel; 16 int32 each
    const int   colB  = tid >> 1;       // 0..127
    const int   halfB = tid & 1;
    const float scl   = scales[nBase + colB];

    const __bf16* gA = x  + (size_t)(mBase + rowA) * INF;
    const int*    gB = qw + (size_t)(nBase + colB) * INF + halfB * 16;

    __bf16* sA = &As[rowA * LDT];
    __bf16* sB = &Bs[colB * LDT + halfB * 16];

    v8f acc[4][4];
#pragma unroll
    for (int mt = 0; mt < 4; ++mt)
#pragma unroll
        for (int nt = 0; nt < 4; ++nt)
#pragma unroll
            for (int j = 0; j < 8; ++j)
                acc[mt][nt][j] = 0.0f;

    // ---- preload first K tile into registers ----
    uint4 a0 = *(const uint4*)(gA + 0);
    uint4 a1 = *(const uint4*)(gA + 8);
    uint4 a2 = *(const uint4*)(gA + 16);
    uint4 a3 = *(const uint4*)(gA + 24);
    int4  q0 = *(const int4*)(gB + 0);
    int4  q1 = *(const int4*)(gB + 4);
    int4  q2 = *(const int4*)(gB + 8);
    int4  q3 = *(const int4*)(gB + 12);

    const int KT = INF / BK;            // 128 iterations
    for (int kt = 0; kt < KT; ++kt) {
        __syncthreads();                // previous compute done reading LDS

        // ---- stage A (raw copy, one row per thread) ----
        *(uint4*)(sA + 0)  = a0;
        *(uint4*)(sA + 8)  = a1;
        *(uint4*)(sA + 16) = a2;
        *(uint4*)(sA + 24) = a3;

        // ---- stage B (dequantize int32 -> bf16) ----
        uint4 b0, b1;
        b0.x = dq2(q0.x, q0.y, scl);  b0.y = dq2(q0.z, q0.w, scl);
        b0.z = dq2(q1.x, q1.y, scl);  b0.w = dq2(q1.z, q1.w, scl);
        b1.x = dq2(q2.x, q2.y, scl);  b1.y = dq2(q2.z, q2.w, scl);
        b1.z = dq2(q3.x, q3.y, scl);  b1.w = dq2(q3.z, q3.w, scl);
        *(uint4*)(sB)     = b0;
        *(uint4*)(sB + 8) = b1;

        // ---- software pipeline: issue next tile's global loads now ----
        if (kt + 1 < KT) {
            const __bf16* pA = gA + (size_t)(kt + 1) * BK;
            const int*    pB = gB + (size_t)(kt + 1) * BK;
            a0 = *(const uint4*)(pA + 0);
            a1 = *(const uint4*)(pA + 8);
            a2 = *(const uint4*)(pA + 16);
            a3 = *(const uint4*)(pA + 24);
            q0 = *(const int4*)(pB + 0);
            q1 = *(const int4*)(pB + 4);
            q2 = *(const int4*)(pB + 8);
            q3 = *(const int4*)(pB + 12);
        }

        __syncthreads();                // LDS tiles ready

        // ---- B fragments first (reused by all 4 mt) ----
        // B 32x16 bf16: lane = N column; lane<16 holds K 0..15, lane>=16 holds 16..31
        v16bf bfrag[4];
#pragma unroll
        for (int nt = 0; nt < 4; ++nt) {
            const __bf16* base = &Bs[(wn * 64 + nt * 16 + laneLow) * LDT + laneHi * 16];
            v8bf lo = *(const v8bf*)(base);
            v8bf hi = *(const v8bf*)(base + 8);
            bfrag[nt] = comb(lo, hi);
        }

        // ---- per-mt: one live A fragment, 4 WMMAs ----
        // A 16x32 bf16: lane<16 holds K 0..15; lane>=16 holds 16..31
#pragma unroll
        for (int mt = 0; mt < 4; ++mt) {
            const __bf16* base = &As[(wm * 64 + mt * 16 + laneLow) * LDT];
            v8bf lo = *(const v8bf*)(base + laneHi * 16);
            v8bf hi = *(const v8bf*)(base + laneHi * 16 + 8);
            v16bf afrag = comb(lo, hi);
#pragma unroll
            for (int nt = 0; nt < 4; ++nt)
                acc[mt][nt] = __builtin_amdgcn_wmma_f32_16x16x32_bf16(
                    false, afrag, false, bfrag[nt],
                    (short)0, acc[mt][nt], false, false);
        }
    }

    // ---- epilogue: out = acc*mul + (bias*mul + add), bf16 store ----
    // C/D layout: lane<16 -> M = j, lane>=16 -> M = 8+j ; N = lane&15
#pragma unroll
    for (int nt = 0; nt < 4; ++nt) {
        const int col = nBase + wn * 64 + nt * 16 + laneLow;
        const float cm = (float)mulv[col];
        const float ca = (float)bias[col] * cm + (float)addv[col];
#pragma unroll
        for (int mt = 0; mt < 4; ++mt) {
            const int row0 = mBase + wm * 64 + mt * 16 + laneHi * 8;
#pragma unroll
            for (int j = 0; j < 8; ++j) {
                float v = acc[mt][nt][j] * cm + ca;
                out[(size_t)(row0 + j) * OUTF + col] = (__bf16)v;
            }
        }
    }
}

extern "C" void kernel_launch(void* const* d_in, const int* in_sizes, int n_in,
                              void* d_out, int out_size, void* d_ws, size_t ws_size,
                              hipStream_t stream) {
    const __bf16* x   = (const __bf16*)d_in[0];
    const int*    qw  = (const int*)d_in[1];
    const float*  sc  = (const float*)d_in[2];
    const __bf16* bs  = (const __bf16*)d_in[3];
    const __bf16* mv  = (const __bf16*)d_in[4];
    const __bf16* av  = (const __bf16*)d_in[5];
    __bf16*       out = (__bf16*)d_out;

    dim3 grid(OUTF / BN, TOKENS / BM);   // (32, 64) = 2048 workgroups
    woq_wmma_kernel<<<grid, 256, 0, stream>>>(x, qw, sc, bs, mv, av, out);
}
